// MultiHeadAttention_9019431321717
// MI455X (gfx1250) — matrix operand order repaired and checked by reading, not compiled, not run on hardware
//
#include <hip/hip_runtime.h>

#define D_MODEL 2048
#define N_HEADS 16
#define HEAD_DIM 128
#define SEQ_L 2048
#define BATCH 2

typedef __attribute__((ext_vector_type(4)))  __bf16 v4bf;
typedef __attribute__((ext_vector_type(8)))  __bf16 v8bf;
typedef __attribute__((ext_vector_type(16))) __bf16 v16bf;
typedef __attribute__((ext_vector_type(4)))  float  v4f;
typedef __attribute__((ext_vector_type(8)))  float  v8f;

__device__ __forceinline__ v16bf cat8(v8bf lo, v8bf hi) {
  return __builtin_shufflevector(lo, hi, 0,1,2,3,4,5,6,7,8,9,10,11,12,13,14,15);
}

__device__ __forceinline__ v8f wmma_bf16(v16bf a, v16bf b, v8f c) {
  // D(f32 16x16) = A(16x32 bf16) * B(32x16 bf16) + C
  return __builtin_amdgcn_wmma_f32_16x16x32_bf16(false, a, false, b, (short)0, c,
                                                 false, false);
}

// ---------------------------------------------------------------------------
// Tiled GEMM: Y[M,N] = X[M,K] @ W[N,K]^T + bias   (torch Linear semantics)
// Block = 256 threads (8 waves). Block tile 128(M) x 128(N), K-chunk 32.
// Wave (widm = wid&3, widn = wid>>2) computes a 32x64 patch: 2 m-tiles x
// 4 n-tiles of 16x16 via WMMA bf16 => 8 WMMAs per K-chunk per wave.
// ---------------------------------------------------------------------------
template <typename TIN, typename TOUT>
__global__ __launch_bounds__(256) void gemm_wmma(const TIN* __restrict__ X,
                                                 const float* __restrict__ W,
                                                 const float* __restrict__ bias,
                                                 TOUT* __restrict__ Y,
                                                 int M, int N, int K) {
  __shared__ __bf16 Xs[128 * 32];
  __shared__ __bf16 Ws[128 * 32];
  const int tid  = threadIdx.x;
  const int lane = tid & 31;
  const int wid  = tid >> 5;
  const int n0 = blockIdx.x * 128;
  const int m0 = blockIdx.y * 128;

  const int r   = lane & 15;
  const int bkA = (lane < 16) ? 0 : 16;  // A frag K-group base
  const int bkB = (lane < 16) ? 0 : 16;  // B frag K-group base
  const int mW  = (wid & 3) * 32;        // wave's m offset in block tile
  const int nW  = (wid >> 2) * 64;       // wave's n offset in block tile

  v8f acc[2][4] = {};

  for (int kt = 0; kt < K; kt += 32) {
    // cooperative global->LDS load with fp32->bf16 conversion
#pragma unroll
    for (int it = 0; it < 4; ++it) {
      int fi  = it * 256 + tid;    // 0..1023 groups of 4 elems
      int row = fi >> 3;           // 0..127
      int c4  = (fi & 7) * 4;
      v4bf xo, wo;
      if constexpr (__is_same(TIN, float)) {
        v4f xv = *(const v4f*)(X + (size_t)(m0 + row) * K + kt + c4);
        xo[0] = (__bf16)xv[0]; xo[1] = (__bf16)xv[1];
        xo[2] = (__bf16)xv[2]; xo[3] = (__bf16)xv[3];
      } else {
        xo = *(const v4bf*)(X + (size_t)(m0 + row) * K + kt + c4);
      }
      v4f wv = *(const v4f*)(W + (size_t)(n0 + row) * K + kt + c4);
      wo[0] = (__bf16)wv[0]; wo[1] = (__bf16)wv[1];
      wo[2] = (__bf16)wv[2]; wo[3] = (__bf16)wv[3];
      *(v4bf*)&Xs[row * 32 + c4] = xo;
      *(v4bf*)&Ws[row * 32 + c4] = wo;
    }
    __syncthreads();

    // A fragments (2 m-tiles of this wave)
    v16bf af[2];
#pragma unroll
    for (int mt = 0; mt < 2; ++mt) {
      const __bf16* ar = &Xs[(mW + mt * 16 + r) * 32 + bkA];
      af[mt] = cat8(*(const v8bf*)ar, *(const v8bf*)(ar + 8));
    }
#pragma unroll
    for (int nt = 0; nt < 4; ++nt) {
      const __bf16* br = &Ws[(nW + nt * 16 + r) * 32 + bkB];
      v16bf bfr = cat8(*(const v8bf*)br, *(const v8bf*)(br + 8));
#pragma unroll
      for (int mt = 0; mt < 2; ++mt)
        acc[mt][nt] = wmma_bf16(af[mt], bfr, acc[mt][nt]);
    }
    __syncthreads();
  }

  // epilogue: C/D layout row = i + 8*(lane>=16), col = lane&15
  const int rhi = (lane >> 4) * 8;
#pragma unroll
  for (int mt = 0; mt < 2; ++mt) {
#pragma unroll
    for (int nt = 0; nt < 4; ++nt) {
      int col  = n0 + nW + nt * 16 + r;
      float bv = bias[col];
#pragma unroll
      for (int i = 0; i < 8; ++i) {
        int row = m0 + mW + mt * 16 + i + rhi;
        float o = acc[mt][nt][i] + bv;
        if constexpr (__is_same(TOUT, float))
          Y[(size_t)row * N + col] = o;
        else
          Y[(size_t)row * N + col] = (__bf16)o;
      }
    }
  }
}

// ---------------------------------------------------------------------------
// RoPE applied in-place to bf16 Q and K buffers laid out [B, L, H*d].
// Thread handles pair (j, j+64) of one (b, l, h).
// ---------------------------------------------------------------------------
__global__ __launch_bounds__(256) void rope_kernel(__bf16* __restrict__ Qb,
                                                   __bf16* __restrict__ Kb) {
  size_t i = (size_t)blockIdx.x * blockDim.x + threadIdx.x;
  int j = (int)(i & 63);
  size_t p = i >> 6;
  int h = (int)(p & 15);  p >>= 4;
  int l = (int)(p & (SEQ_L - 1));
  int b = (int)(p >> 11);
  // inv_freq = base^(-j/64);  log(10000)/64 = 0.14391156831...
  float ang = (float)l * __expf(-(float)j * 0.14391156831212787f);
  float s, c;
  __sincosf(ang, &s, &c);
  size_t base = ((size_t)(b * SEQ_L + l)) * D_MODEL + h * HEAD_DIM + j;

  float q0 = (float)Qb[base], q1 = (float)Qb[base + 64];
  Qb[base]      = (__bf16)(q0 * c - q1 * s);
  Qb[base + 64] = (__bf16)(q1 * c + q0 * s);

  float k0 = (float)Kb[base], k1 = (float)Kb[base + 64];
  Kb[base]      = (__bf16)(k0 * c - k1 * s);
  Kb[base + 64] = (__bf16)(k1 * c + k0 * s);
}

// ---------------------------------------------------------------------------
// Causal flash attention, bf16 WMMA, fp32 online softmax.
// Block = 64 threads (2 waves); each wave owns a 16-query tile and streams
// 32-key blocks. K tile staged with CDNA5 async global->LDS (ASYNCcnt);
// V staged through VGPRs (transposed store). LDS per wave: 17KB.
// ---------------------------------------------------------------------------
__global__ __launch_bounds__(64) void flash_attn(const __bf16* __restrict__ Qb,
                                                 const __bf16* __restrict__ Kb,
                                                 const __bf16* __restrict__ Vb,
                                                 float* __restrict__ AO) {
  __shared__ __bf16 smem[2 * (32 * 128 + 128 * 32 + 16 * 32)];
  const int lane = threadIdx.x & 31;
  const int wid  = threadIdx.x >> 5;
  __bf16* Ks = smem + wid * 8704;
  __bf16* Vt = Ks + 4096;
  __bf16* Ps = Vt + 4096;

  const int b    = blockIdx.x >> 4;
  const int h    = blockIdx.x & 15;
  const int q0   = blockIdx.y * 32 + wid * 16;
  const int r    = lane & 15;
  const int hi16 = lane >> 4;            // 0 / 1
  const int bkA  = hi16 ? 16 : 0;
  const int bkB  = hi16 ? 16 : 0;

  // Q fragments (A layout), straight from global bf16
  v16bf qf[4];
  {
    const __bf16* qrow =
        Qb + ((size_t)(b * SEQ_L + q0 + r)) * D_MODEL + h * HEAD_DIM;
#pragma unroll
    for (int c = 0; c < 4; ++c) {
      v8bf lo = *(const v8bf*)(qrow + c * 32 + bkA);
      v8bf hv = *(const v8bf*)(qrow + c * 32 + bkA + 8);
      qf[c] = cat8(lo, hv);
    }
  }

  float m_i[8], l_i[8];
  v8f acc[8] = {};
#pragma unroll
  for (int i = 0; i < 8; ++i) { m_i[i] = -1e30f; l_i[i] = 0.0f; }

  const int nkb = (q0 + 16 + 31) >> 5;   // causal: keys 0 .. q0+15
  for (int kb = 0; kb < nkb; ++kb) {
    const int k0 = kb * 32;

    // prefetch next key block (global_prefetch_b8)
    if (kb + 1 < nkb) {
      const __bf16* nk =
          Kb + (size_t)(b * SEQ_L + k0 + 32 + (lane >> 2)) * D_MODEL +
          h * HEAD_DIM + (lane & 3) * 32;
      __builtin_prefetch(nk, 0, 1);
      const __bf16* nv =
          Vb + (size_t)(b * SEQ_L + k0 + 32 + (lane >> 2)) * D_MODEL +
          h * HEAD_DIM + (lane & 3) * 32;
      __builtin_prefetch(nv, 0, 1);
    }

    // intra-wave cooperative staging: K via async global->LDS, V transposed
#pragma unroll 4
    for (int it = 0; it < 16; ++it) {
      int f   = it * 32 + lane;          // 512 v8bf chunks
      int row = f >> 4;                  // key 0..31
      int c8  = (f & 15) * 8;            // dim
      size_t g = (size_t)(b * SEQ_L + k0 + row) * D_MODEL + h * HEAD_DIM + c8;
      // K: 16B/lane async copy to LDS, tracked by ASYNCcnt
      unsigned klds = (unsigned)(size_t)&Ks[row * 128 + c8];
      asm volatile("global_load_async_to_lds_b128 %0, %1, off"
                   :: "v"(klds), "v"(Kb + g) : "memory");
      // V: through VGPRs, stored transposed [dim][key]
      v8bf vv = *(const v8bf*)(Vb + g);
#pragma unroll
      for (int j = 0; j < 8; ++j) Vt[(c8 + j) * 32 + row] = vv[j];
    }
    asm volatile("s_wait_asynccnt 0" ::: "memory");
    asm volatile("s_wait_dscnt 0" ::: "memory");

    // S = Q K^T : two 16x16 score tiles, K accumulated over d=128
    v8f s0 = {}, s1 = {};
#pragma unroll
    for (int c = 0; c < 4; ++c) {
      const __bf16* k0p = &Ks[r * 128 + c * 32 + bkB];
      v16bf kf0 = cat8(*(const v8bf*)k0p, *(const v8bf*)(k0p + 8));
      const __bf16* k1p = &Ks[(r + 16) * 128 + c * 32 + bkB];
      v16bf kf1 = cat8(*(const v8bf*)k1p, *(const v8bf*)(k1p + 8));
      s0 = wmma_bf16(qf[c], kf0, s0);
      s1 = wmma_bf16(qf[c], kf1, s1);
    }
    const float scale = 0.08838834764831845f;  // 1/sqrt(128)
#pragma unroll
    for (int i = 0; i < 8; ++i) { s0[i] *= scale; s1[i] *= scale; }

    if (k0 + 31 > q0) {  // diagonal blocks: causal mask
#pragma unroll
      for (int i = 0; i < 8; ++i) {
        int q = q0 + i + hi16 * 8;
        if (k0 + r > q)      s0[i] = -1e30f;
        if (k0 + 16 + r > q) s1[i] = -1e30f;
      }
    }

    // online softmax (row = i + 8*hi16 lives in 16-lane group of one VGPR)
#pragma unroll
    for (int i = 0; i < 8; ++i) {
      float v = fmaxf(s0[i], s1[i]);
      v = fmaxf(v, __shfl_xor(v, 1, 32));
      v = fmaxf(v, __shfl_xor(v, 2, 32));
      v = fmaxf(v, __shfl_xor(v, 4, 32));
      v = fmaxf(v, __shfl_xor(v, 8, 32));
      float mn = fmaxf(m_i[i], v);
      float alpha = __expf(m_i[i] - mn);
      m_i[i] = mn;
      float p0 = __expf(s0[i] - mn);
      float p1 = __expf(s1[i] - mn);
      float rs = p0 + p1;
      rs += __shfl_xor(rs, 1, 32);
      rs += __shfl_xor(rs, 2, 32);
      rs += __shfl_xor(rs, 4, 32);
      rs += __shfl_xor(rs, 8, 32);
      l_i[i] = l_i[i] * alpha + rs;
#pragma unroll
      for (int t = 0; t < 8; ++t) acc[t][i] *= alpha;
      int prow = i + hi16 * 8;
      Ps[prow * 32 + r]      = (__bf16)p0;
      Ps[prow * 32 + 16 + r] = (__bf16)p1;
    }
    asm volatile("s_wait_dscnt 0" ::: "memory");

    // P (16x32) as one A fragment; acc += P @ V
    const __bf16* pp = &Ps[r * 32 + bkA];
    v16bf pf = cat8(*(const v8bf*)pp, *(const v8bf*)(pp + 8));
#pragma unroll
    for (int t = 0; t < 8; ++t) {
      const __bf16* vp = &Vt[(t * 16 + r) * 32 + bkB];
      v16bf vf = cat8(*(const v8bf*)vp, *(const v8bf*)(vp + 8));
      acc[t] = wmma_bf16(pf, vf, acc[t]);
    }
    // drain DS reads before next iteration's async/DS writes can land
    asm volatile("s_wait_dscnt 0" ::: "memory");
  }

  // out = acc / l, written fp32 into [B, L, H*d]
#pragma unroll
  for (int t = 0; t < 8; ++t) {
#pragma unroll
    for (int i = 0; i < 8; ++i) {
      int row = q0 + i + hi16 * 8;
      int col = h * HEAD_DIM + t * 16 + r;
      AO[(size_t)(b * SEQ_L + row) * D_MODEL + col] = acc[t][i] / l_i[i];
    }
  }
}

// ---------------------------------------------------------------------------
extern "C" void kernel_launch(void* const* d_in, const int* in_sizes, int n_in,
                              void* d_out, int out_size, void* d_ws,
                              size_t ws_size, hipStream_t stream) {
  (void)in_sizes; (void)n_in; (void)out_size; (void)ws_size;
  const float* X  = (const float*)d_in[0];
  const float* Wq = (const float*)d_in[1];
  const float* bq = (const float*)d_in[2];
  const float* Wk = (const float*)d_in[3];
  const float* bk = (const float*)d_in[4];
  const float* Wv = (const float*)d_in[5];
  const float* bv = (const float*)d_in[6];
  const float* Wo = (const float*)d_in[7];
  const float* bo = (const float*)d_in[8];
  float* OUT = (float*)d_out;

  const int M = BATCH * SEQ_L;      // 4096
  const int N = D_MODEL, K = D_MODEL;
  const size_t elems = (size_t)M * D_MODEL;  // 8388608

  __bf16* Qb = (__bf16*)d_ws;
  __bf16* Kb = Qb + elems;
  __bf16* Vb = Kb + elems;
  float*  AO = (float*)(Vb + elems);

  dim3 gblk(256);
  dim3 ggrid(N / 128, M / 128);     // (16, 32)
  gemm_wmma<float, __bf16><<<ggrid, gblk, 0, stream>>>(X, Wq, bq, Qb, M, N, K);
  gemm_wmma<float, __bf16><<<ggrid, gblk, 0, stream>>>(X, Wk, bk, Kb, M, N, K);
  gemm_wmma<float, __bf16><<<ggrid, gblk, 0, stream>>>(X, Wv, bv, Vb, M, N, K);

  size_t nrope = (size_t)BATCH * SEQ_L * N_HEADS * 64;  // 4194304
  rope_kernel<<<dim3((unsigned)(nrope / 256)), dim3(256), 0, stream>>>(Qb, Kb);

  flash_attn<<<dim3(BATCH * N_HEADS, SEQ_L / 32), dim3(64), 0, stream>>>(
      Qb, Kb, Vb, AO);

  gemm_wmma<float, float><<<ggrid, gblk, 0, stream>>>(AO, Wo, bo, OUT, M, N, K);
}
